// LlamaMLP_59124519796893
// MI455X (gfx1250) — 8-bit matrix operand order repaired and checked by reading, compiled, not run on hardware
//
#include <hip/hip_runtime.h>
#include <math.h>

typedef __attribute__((ext_vector_type(8))) int v8i;

#define T_DIM 2048
#define H_DIM 4096
#define I_DIM 11008

__device__ __forceinline__ v8i wmma_iu8(v8i a, v8i b, v8i c) {
    return __builtin_amdgcn_wmma_i32_16x16x64_iu8(true, a, true, b, c, false, false);
}

__device__ __forceinline__ v8i load_a_tile(const signed char* __restrict__ base,
                                           int row, int K, int kbase, int hi) {
    const signed char* p = base + (size_t)row * K + kbase + hi * 16;
    int2 d0 = *(const int2*)(p + 0);
    int2 d1 = *(const int2*)(p + 8);
    int2 d2 = *(const int2*)(p + 32);
    int2 d3 = *(const int2*)(p + 40);
    v8i a;
    a[0] = d0.x; a[1] = d0.y;
    a[2] = d1.x; a[3] = d1.y;
    a[4] = d2.x; a[5] = d2.y;
    a[6] = d3.x; a[7] = d3.y;
    return a;
}

__device__ __forceinline__ v8i load_b_tile(const signed char* __restrict__ base,
                                           int col, int K, int kbase, int khalf) {
    const signed char* p = base + (size_t)col * K + kbase + khalf * 16;
    int4 q0 = *(const int4*)(p + 0);
    int4 q1 = *(const int4*)(p + 32);
    v8i b;
    b[0] = q0.x; b[1] = q0.y; b[2] = q0.z; b[3] = q0.w;
    b[4] = q1.x; b[5] = q1.y; b[6] = q1.z; b[7] = q1.w;
    return b;
}

__device__ __forceinline__ float silu_mul(float g, float u) {
    return g * u * __builtin_amdgcn_rcpf(1.0f + __expf(-g));
}

__global__ __launch_bounds__(256)
void gemm1_gate_up_silu(const signed char* __restrict__ x_q,
                        const float* __restrict__ scale_x,
                        const signed char* __restrict__ w_gu,
                        const float* __restrict__ s_gu,
                        float* __restrict__ a_out,
                        unsigned int* __restrict__ absmax) {
    const int lane = threadIdx.x & 31;
    const int wave = threadIdx.x >> 5;
    const int lo = lane & 15;
    const int hi = lane >> 4;
    const int m0 = blockIdx.x * 32;
    const int cg = blockIdx.y * 256 + wave * 32;

    v8i g00 = {}, g01 = {}, g10 = {}, g11 = {};
    v8i u00 = {}, u01 = {}, u10 = {}, u11 = {};

    for (int k = 0; k < H_DIM; k += 64) {
        v8i a0  = load_a_tile(x_q, m0 + lo,        H_DIM, k, hi);
        v8i a1  = load_a_tile(x_q, m0 + 16 + lo,   H_DIM, k, hi);
        v8i bg0 = load_b_tile(w_gu, cg + lo,               H_DIM, k, hi);
        v8i bg1 = load_b_tile(w_gu, cg + 16 + lo,          H_DIM, k, hi);
        v8i bu0 = load_b_tile(w_gu, cg + I_DIM + lo,       H_DIM, k, hi);
        v8i bu1 = load_b_tile(w_gu, cg + I_DIM + 16 + lo,  H_DIM, k, hi);
        g00 = wmma_iu8(a0, bg0, g00);
        g01 = wmma_iu8(a0, bg1, g01);
        g10 = wmma_iu8(a1, bg0, g10);
        g11 = wmma_iu8(a1, bg1, g11);
        u00 = wmma_iu8(a0, bu0, u00);
        u01 = wmma_iu8(a0, bu1, u01);
        u10 = wmma_iu8(a1, bu0, u10);
        u11 = wmma_iu8(a1, bu1, u11);
    }

    const int n0 = cg + lo;
    const int n1 = cg + 16 + lo;
    const float sg0 = s_gu[n0], sg1 = s_gu[n1];
    const float su0 = s_gu[n0 + I_DIM], su1 = s_gu[n1 + I_DIM];

#pragma unroll
    for (int mt = 0; mt < 2; ++mt) {
        const v8i* G0 = mt ? &g10 : &g00;
        const v8i* G1 = mt ? &g11 : &g01;
        const v8i* U0 = mt ? &u10 : &u00;
        const v8i* U1 = mt ? &u11 : &u01;
#pragma unroll
        for (int vi = 0; vi < 8; ++vi) {
            const int row = m0 + mt * 16 + vi + 8 * hi;
            const float sx = scale_x[row];
            const float ga = (float)((*G0)[vi]) * sx * sg0;
            const float gb = (float)((*G1)[vi]) * sx * sg1;
            const float ua = (float)((*U0)[vi]) * sx * su0;
            const float ub = (float)((*U1)[vi]) * sx * su1;
            const float av0 = silu_mul(ga, ua);
            const float av1 = silu_mul(gb, ub);
            float* rowp = a_out + (size_t)row * I_DIM;
            rowp[n0] = av0;
            rowp[n1] = av1;

            float m = fmaxf(fabsf(av0), fabsf(av1));
            m = fmaxf(m, __shfl_xor(m, 1, 32));
            m = fmaxf(m, __shfl_xor(m, 2, 32));
            m = fmaxf(m, __shfl_xor(m, 4, 32));
            m = fmaxf(m, __shfl_xor(m, 8, 32));
            if (lo == 0)
                atomicMax(&absmax[row], __float_as_uint(m));
        }
    }
}

__global__ __launch_bounds__(256)
void quantize_act(const float* __restrict__ a,
                  const unsigned int* __restrict__ absmax,
                  signed char* __restrict__ a_q) {
    const size_t idx = ((size_t)blockIdx.x * blockDim.x + threadIdx.x) * 4;
    const int row = (int)(idx / I_DIM);
    const float s = __uint_as_float(absmax[row]) * (1.0f / 127.0f);
    const float inv = (s > 0.0f) ? __builtin_amdgcn_rcpf(s) : 0.0f;

    const float4 v = *(const float4*)(a + idx);
    float qx = fminf(fmaxf(rintf(v.x * inv), -127.0f), 127.0f);
    float qy = fminf(fmaxf(rintf(v.y * inv), -127.0f), 127.0f);
    float qz = fminf(fmaxf(rintf(v.z * inv), -127.0f), 127.0f);
    float qw = fminf(fmaxf(rintf(v.w * inv), -127.0f), 127.0f);

    char4 q;
    q.x = (signed char)(int)qx;
    q.y = (signed char)(int)qy;
    q.z = (signed char)(int)qz;
    q.w = (signed char)(int)qw;
    *(char4*)(a_q + idx) = q;
}

__global__ __launch_bounds__(256)
void gemm2_down(const signed char* __restrict__ a_q,
                const unsigned int* __restrict__ absmax,
                const signed char* __restrict__ w_down,
                const float* __restrict__ s_down,
                float* __restrict__ out) {
    const int lane = threadIdx.x & 31;
    const int wave = threadIdx.x >> 5;
    const int lo = lane & 15;
    const int hi = lane >> 4;
    const int m0 = blockIdx.x * 32;
    const int h0 = blockIdx.y * 256 + wave * 32;

    v8i c00 = {}, c01 = {}, c10 = {}, c11 = {};

    for (int k = 0; k < I_DIM; k += 64) {
        v8i a0 = load_a_tile(a_q, m0 + lo,      I_DIM, k, hi);
        v8i a1 = load_a_tile(a_q, m0 + 16 + lo, I_DIM, k, hi);
        v8i b0 = load_b_tile(w_down, h0 + lo,      I_DIM, k, hi);
        v8i b1 = load_b_tile(w_down, h0 + 16 + lo, I_DIM, k, hi);
        c00 = wmma_iu8(a0, b0, c00);
        c01 = wmma_iu8(a0, b1, c01);
        c10 = wmma_iu8(a1, b0, c10);
        c11 = wmma_iu8(a1, b1, c11);
    }

    const int n0 = h0 + lo;
    const int n1 = h0 + 16 + lo;
    const float sd0 = s_down[n0];
    const float sd1 = s_down[n1];

#pragma unroll
    for (int mt = 0; mt < 2; ++mt) {
        const v8i* C0 = mt ? &c10 : &c00;
        const v8i* C1 = mt ? &c11 : &c01;
#pragma unroll
        for (int vi = 0; vi < 8; ++vi) {
            const int row = m0 + mt * 16 + vi + 8 * hi;
            const float sa = __uint_as_float(absmax[row]) * (1.0f / 127.0f);
            float* rowp = out + (size_t)row * H_DIM;
            rowp[n0] = (float)((*C0)[vi]) * sa * sd0;
            rowp[n1] = (float)((*C1)[vi]) * sa * sd1;
        }
    }
}

extern "C" void kernel_launch(void* const* d_in, const int* in_sizes, int n_in,
                              void* d_out, int out_size, void* d_ws, size_t ws_size,
                              hipStream_t stream) {
    const signed char* x_q     = (const signed char*)d_in[0];
    const float*       scale_x = (const float*)d_in[1];
    const signed char* w_gu    = (const signed char*)d_in[2];
    const float*       s_gu    = (const float*)d_in[3];
    const signed char* w_down  = (const signed char*)d_in[4];
    const float*       s_down  = (const float*)d_in[5];
    float*             out     = (float*)d_out;

    char* ws = (char*)d_ws;
    unsigned int* absmax = (unsigned int*)ws;
    float*        a_f    = (float*)(ws + 8192);
    signed char*  a_q    = (signed char*)(ws + 8192 + (size_t)T_DIM * I_DIM * 4);

    hipMemsetAsync(absmax, 0, T_DIM * sizeof(unsigned int), stream);

    dim3 blk(256);

    dim3 g1(T_DIM / 32, I_DIM / 256);
    gemm1_gate_up_silu<<<g1, blk, 0, stream>>>(x_q, scale_x, w_gu, s_gu, a_f, absmax);

    size_t nq = (size_t)T_DIM * I_DIM / 4;
    quantize_act<<<(unsigned)((nq + 255) / 256), blk, 0, stream>>>(a_f, absmax, a_q);

    dim3 g2(T_DIM / 32, H_DIM / 256);
    gemm2_down<<<g2, blk, 0, stream>>>(a_q, absmax, w_down, s_down, out);
}
